// Model_39676907885160
// MI455X (gfx1250) — matrix operand order repaired and checked by reading, not compiled, not run on hardware
//
#include <hip/hip_runtime.h>
#include <hip/hip_bf16.h>

// ---------------------------------------------------------------------------
// Flash-attention forward for MI455X (gfx1250, wave32, WMMA bf16 16x16x32).
// B=8, SQ=SK=4096, D=128. Compute-bound (AI ~1000 FLOP/B) -> bf16 WMMA path.
// Software-pipelined LDS fragment loads to avoid per-WMMA dscnt stalls.
// ---------------------------------------------------------------------------

typedef __bf16 bf16_t;
typedef __attribute__((ext_vector_type(16))) __bf16 v16bf;
typedef __attribute__((ext_vector_type(8)))  __bf16 v8bf;
typedef __attribute__((ext_vector_type(4)))  __bf16 v4bf;
typedef __attribute__((ext_vector_type(8)))  float  v8f;
typedef __attribute__((ext_vector_type(4)))  float  v4f;

#define B_    8
#define SQ_   4096
#define SK_   4096
#define D_    128

#define BQ    128          // queries per workgroup
#define BK    64           // keys per LDS tile
#define NWAVE 8
#define KPITCH 136         // Ks row pitch (bf16): 272B = 17*16B -> aligned + bank-spread
#define VPITCH 72          // Vt row pitch (bf16): 144B = 9*16B  -> aligned + bank-spread
#define PPITCH 72          // P staging row pitch

__device__ __forceinline__ v8f v8f_zero() {
    v8f z = {0.f, 0.f, 0.f, 0.f, 0.f, 0.f, 0.f, 0.f};
    return z;
}

__device__ __forceinline__ unsigned wang_hash(unsigned h) {
    h = (h ^ 61u) ^ (h >> 16);
    h *= 9u;
    h ^= h >> 4;
    h *= 0x27d4eb2du;
    h ^= h >> 15;
    return h;
}

// Load one 16-wide bf16 WMMA operand (two 16B LDS reads) from `p`.
__device__ __forceinline__ v16bf load_frag16(const bf16_t* p, int second_off) {
    v8bf b0 = *(const v8bf*)p;
    v8bf b1 = *(const v8bf*)(p + second_off);
    v16bf f;
    #pragma unroll
    for (int e = 0; e < 8; ++e) { f[e] = b0[e]; f[8 + e] = b1[e]; }
    return f;
}

__device__ __forceinline__ v8f wmma_bf16(v16bf a, v16bf b, v8f c) {
    return __builtin_amdgcn_wmma_f32_16x16x32_bf16(
        false, a, false, b, (short)0, c, false, false);
}

__global__ __launch_bounds__(256) void fattn_fwd(
    const float* __restrict__ Q, const float* __restrict__ K,
    const float* __restrict__ V, const float* __restrict__ inv_scale_p,
    const float* __restrict__ pdrop_p, const float* __restrict__ mdrop_p,
    float* __restrict__ Out)
{
    __shared__ bf16_t Ks[BK * KPITCH];            // K tile, row-major [key][d]
    __shared__ bf16_t Vt[D_ * VPITCH];            // V tile, transposed [d][key]
    __shared__ bf16_t Ps[NWAVE * 16 * PPITCH];    // per-wave P staging [row][key]

    const int tid  = threadIdx.x;
    const int wave = tid >> 5;
    const int lane = tid & 31;
    const int hf   = lane >> 4;        // which 16-lane half
    const int l16  = lane & 15;
    const int b    = blockIdx.y;
    const int q0   = blockIdx.x * BQ;
    const int qlo  = q0 + wave * 16;   // this wave's 16 query rows

    const float scale       = 1.0f / inv_scale_p[0];
    const float p           = pdrop_p[0];
    const float m           = mdrop_p[0];
    const unsigned keep12   = (unsigned)((1.0f - p) * 4096.0f);  // 12-bit threshold
    const float dscale      = 1.0f / ((1.0f - p) * (1.0f - m));

    // ---- Q A-fragments (16x32 bf16 layout), pre-scaled by 1/sqrt(D) ----
    // lane<16: K = {0..15}; lane>=16: K = {16..31}
    v16bf qf[4];
    {
        const float* qrow = Q + ((size_t)b * SQ_ + qlo + l16) * D_;
        const int b0 = hf ? 16 : 0;
        #pragma unroll
        for (int c = 0; c < 4; ++c) {
            const float* p0 = qrow + 32 * c + b0;
            const float* p1 = p0 + 8;
            v4f x0 = *(const v4f*)(p0);
            v4f x1 = *(const v4f*)(p0 + 4);
            v4f y0 = *(const v4f*)(p1);
            v4f y1 = *(const v4f*)(p1 + 4);
            #pragma unroll
            for (int e = 0; e < 4; ++e) {
                qf[c][e]      = (bf16_t)(x0[e] * scale);
                qf[c][4 + e]  = (bf16_t)(x1[e] * scale);
                qf[c][8 + e]  = (bf16_t)(y0[e] * scale);
                qf[c][12 + e] = (bf16_t)(y1[e] * scale);
            }
        }
    }

    // ---- running state: O accumulators (8 d-tiles), row max, row sum ----
    v8f acc[8];
    #pragma unroll
    for (int t = 0; t < 8; ++t) acc[t] = v8f_zero();
    float mrow[8], lrow[8];
    #pragma unroll
    for (int v = 0; v < 8; ++v) { mrow[v] = -1e30f; lrow[v] = 0.0f; }

    // LDS bases reused every iteration
    const bf16_t* kfrag_base = &Ks[l16 * KPITCH + (hf ? 16 : 0)];
    const bf16_t* vfrag_base = &Vt[l16 * VPITCH + (hf ? 16 : 0)];
    const bf16_t* pfrag_base = &Ps[(wave * 16 + l16) * PPITCH + (hf ? 16 : 0)];
    bf16_t*       pstage     = &Ps[(wave * 16 + hf * 8) * PPITCH + l16];

    for (int kt = 0; kt < SK_; kt += BK) {
        __syncthreads();   // previous-iteration readers done before overwrite
        // ---- cooperative load: K tile (row-major) + V tile (transposed) ----
        {
            const float* Kb = K + ((size_t)b * SK_ + kt) * D_;
            const float* Vb = V + ((size_t)b * SK_ + kt) * D_;
            #pragma unroll
            for (int i = 0; i < 8; ++i) {
                int g   = tid + i * 256;        // 2048 float4 groups = 64x128
                int key = g >> 5;
                int d4  = (g & 31) << 2;
                v4f kv = *(const v4f*)(Kb + key * D_ + d4);
                v4bf kb;
                kb.x = (bf16_t)kv.x; kb.y = (bf16_t)kv.y;
                kb.z = (bf16_t)kv.z; kb.w = (bf16_t)kv.w;
                *(v4bf*)(&Ks[key * KPITCH + d4]) = kb;
                v4f vv = *(const v4f*)(Vb + key * D_ + d4);
                Vt[(d4 + 0) * VPITCH + key] = (bf16_t)vv.x;
                Vt[(d4 + 1) * VPITCH + key] = (bf16_t)vv.y;
                Vt[(d4 + 2) * VPITCH + key] = (bf16_t)vv.z;
                Vt[(d4 + 3) * VPITCH + key] = (bf16_t)vv.w;
            }
        }
        __syncthreads();

        // prefetch next key tile into L2 (gfx1250 global_prefetch_b8 path)
        if (kt + BK < SK_) {
            const float* nK = K + ((size_t)b * SK_ + kt + BK) * D_ + tid * 32;
            const float* nV = V + ((size_t)b * SK_ + kt + BK) * D_ + tid * 32;
            __builtin_prefetch(nK, 0, 1);
            __builtin_prefetch(nV, 0, 1);
        }

        // ---- S = Q * K^T : 4 column tiles x 4 K-chunks, pipelined frags ----
        v8f S[4];
        v16bf kfr[2][4];
        #pragma unroll
        for (int cc = 0; cc < 4; ++cc)
            kfr[0][cc] = load_frag16(kfrag_base + cc * 32, 8);
        #pragma unroll
        for (int j = 0; j < 4; ++j) {
            if (j < 3) {            // issue next tile's 8 ds_load_b128 now
                const bf16_t* nb = kfrag_base + (j + 1) * 16 * KPITCH;
                #pragma unroll
                for (int cc = 0; cc < 4; ++cc)
                    kfr[(j + 1) & 1][cc] = load_frag16(nb + cc * 32, 8);
            }
            v8f c = v8f_zero();
            #pragma unroll
            for (int cc = 0; cc < 4; ++cc)
                c = wmma_bf16(qf[cc], kfr[j & 1][cc], c);
            S[j] = c;
        }

        // ---- online softmax (row reductions stay inside 16-lane halves) ----
        #pragma unroll
        for (int v = 0; v < 8; ++v) {
            float mx = fmaxf(fmaxf(S[0][v], S[1][v]), fmaxf(S[2][v], S[3][v]));
            mx = fmaxf(mx, __shfl_xor(mx, 1));
            mx = fmaxf(mx, __shfl_xor(mx, 2));
            mx = fmaxf(mx, __shfl_xor(mx, 4));
            mx = fmaxf(mx, __shfl_xor(mx, 8));
            float mnew  = fmaxf(mrow[v], mx);
            float alpha = __expf(mrow[v] - mnew);
            mrow[v] = mnew;
            float rs = 0.0f;
            #pragma unroll
            for (int j = 0; j < 4; ++j) {
                float e = __expf(S[j][v] - mnew);
                S[j][v] = e;
                rs += e;
            }
            rs += __shfl_xor(rs, 1);
            rs += __shfl_xor(rs, 2);
            rs += __shfl_xor(rs, 4);
            rs += __shfl_xor(rs, 8);
            lrow[v] = lrow[v] * alpha + rs;          // sum BEFORE dropout
            #pragma unroll
            for (int t = 0; t < 8; ++t) acc[t][v] *= alpha;
        }

        // ---- dropout + stage P as bf16 in wave-private LDS (C -> A relayout)
        //      one Wang hash -> two 12-bit keep decisions (adjacent j tiles)
        #pragma unroll
        for (int v = 0; v < 8; ++v) {
            int row = v + hf * 8;
            unsigned qidx = (unsigned)(b * SQ_ + qlo + row) << 12;
            bf16_t* prow = pstage + v * PPITCH;
            #pragma unroll
            for (int j = 0; j < 4; j += 2) {
                unsigned h = wang_hash(qidx | (unsigned)(kt + j * 16 + l16));
                float p0 = ((h & 0xFFFu) < keep12)         ? S[j][v]     * dscale : 0.0f;
                float p1 = (((h >> 12) & 0xFFFu) < keep12) ? S[j + 1][v] * dscale : 0.0f;
                prow[j * 16]       = (bf16_t)p0;
                prow[(j + 1) * 16] = (bf16_t)p1;
            }
        }
        // DS ops are in-order per wave: staging stores complete before loads.

        // ---- O += P * V : 2 K-chunks x 8 d-tiles, pipelined V frags ----
        #pragma unroll
        for (int c2 = 0; c2 < 2; ++c2) {
            // A 16x32 layout: contiguous run of 16 at kbase + 16*hf
            v16bf pf = load_frag16(pfrag_base + c2 * 32, 8);
            v16bf vfr[2];
            vfr[0] = load_frag16(vfrag_base + c2 * 32, 8);
            #pragma unroll
            for (int t = 0; t < 8; ++t) {
                if (t < 7)
                    vfr[(t + 1) & 1] =
                        load_frag16(vfrag_base + (t + 1) * 16 * VPITCH + c2 * 32, 8);
                acc[t] = wmma_bf16(pf, vfr[t & 1], acc[t]);
            }
        }
    }

    // ---- epilogue: O / l ----
    #pragma unroll
    for (int v = 0; v < 8; ++v) {
        float inv_l = 1.0f / lrow[v];
        int row = v + hf * 8;
        float* orow = Out + ((size_t)b * SQ_ + qlo + row) * D_ + l16;
        #pragma unroll
        for (int t = 0; t < 8; ++t) orow[t * 16] = acc[t][v] * inv_l;
    }
}

extern "C" void kernel_launch(void* const* d_in, const int* in_sizes, int n_in,
                              void* d_out, int out_size, void* d_ws, size_t ws_size,
                              hipStream_t stream) {
    (void)in_sizes; (void)n_in; (void)out_size; (void)d_ws; (void)ws_size;
    const float* q  = (const float*)d_in[0];
    const float* k  = (const float*)d_in[1];
    const float* v  = (const float*)d_in[2];
    const float* is = (const float*)d_in[3];
    const float* dp = (const float*)d_in[4];
    const float* dm = (const float*)d_in[5];
    float* out = (float*)d_out;
    dim3 grid(SQ_ / BQ, B_);
    fattn_fwd<<<grid, 256, 0, stream>>>(q, k, v, is, dp, dm, out);
}
